// GCNWithAttention_45801531244894
// MI455X (gfx1250) — compile-verified, matrix operand order checked by reading, not run on hardware
//
#include <hip/hip_runtime.h>
#include <hip/hip_bf16.h>
#include <math.h>

#define N_NODES 50000
#define N_EDGES 800000
#define DIM_IN  128
#define DIM_HID 64
#define DIM_OUT 32
#define N_HEADS 2

typedef _Float16 half_t;
typedef __attribute__((ext_vector_type(16))) _Float16 v16h;
typedef __attribute__((ext_vector_type(8)))  float    v8f;

// ---------------------------------------------------------------------------
// Utility kernels
// ---------------------------------------------------------------------------
__global__ void fill_kernel(float* __restrict__ p, int n, float v) {
  int i = blockIdx.x * blockDim.x + threadIdx.x;
  if (i < n) p[i] = v;
}

__global__ void deg_kernel(const int* __restrict__ ei, float* __restrict__ deg) {
  int e = blockIdx.x * blockDim.x + threadIdx.x;
  if (e < N_EDGES) atomicAdd(&deg[ei[N_EDGES + e]], 1.0f);
}

__global__ void dinv_kernel(const float* __restrict__ deg, float* __restrict__ dinv) {
  int i = blockIdx.x * blockDim.x + threadIdx.x;
  if (i < N_NODES) dinv[i] = rsqrtf(deg[i] + 1.0f);  // +1 for self loop
}

// ---------------------------------------------------------------------------
// WMMA GEMM: C[M x Kout] = A[M x Kin] * W[Kin x Kout]
// f32 -> f16 inputs, f32 accumulate via v_wmma_f32_16x16x32_f16.
// M multiple of 16, Kin multiple of 32, Kout multiple of 16.
// One wave per 16x16 C tile; guards are wave-uniform so EXEC is all-ones.
// A-matrix lane layout (16-bit, 16x32): lanes 0-15 row M=lane, halves [0..7]=K
// kb+0..7, [8..15]=K kb+16..23 with kb=0; lanes 16-31 same rows with kb=8.
// B-matrix mirrors A with N in place of M (column-per-lane).
// C/D f32: lane&15 = N, VGPR r -> M = r + (lane<16 ? 0 : 8).
// ---------------------------------------------------------------------------
__global__ void wmma_gemm_f16(const float* __restrict__ A, const float* __restrict__ W,
                              float* __restrict__ C, int M, int Kin, int Kout) {
  const int lane  = threadIdx.x & 31;
  const int wave  = blockIdx.x * (blockDim.x >> 5) + (threadIdx.x >> 5);
  const int tilesN = Kout >> 4;
  const int tileM  = wave / tilesN;
  const int tileN  = wave % tilesN;
  if (tileM * 16 >= M) return;                 // wave-uniform
  const int row = tileM * 16 + (lane & 15);
  const int col = tileN * 16 + (lane & 15);
  const int kb  = (lane < 16) ? 0 : 8;

  v8f acc = {};
  for (int k0 = 0; k0 < Kin; k0 += 32) {
    const float* ap = A + (size_t)row * Kin + k0 + kb;
    float4 a0 = *(const float4*)(ap);
    float4 a1 = *(const float4*)(ap + 4);
    float4 a2 = *(const float4*)(ap + 16);
    float4 a3 = *(const float4*)(ap + 20);
    v16h av, bv;
    av[0]  = (half_t)a0.x; av[1]  = (half_t)a0.y; av[2]  = (half_t)a0.z; av[3]  = (half_t)a0.w;
    av[4]  = (half_t)a1.x; av[5]  = (half_t)a1.y; av[6]  = (half_t)a1.z; av[7]  = (half_t)a1.w;
    av[8]  = (half_t)a2.x; av[9]  = (half_t)a2.y; av[10] = (half_t)a2.z; av[11] = (half_t)a2.w;
    av[12] = (half_t)a3.x; av[13] = (half_t)a3.y; av[14] = (half_t)a3.z; av[15] = (half_t)a3.w;
#pragma unroll
    for (int h = 0; h < 16; ++h) {
      int kk = k0 + kb + ((h < 8) ? h : (h + 8));
      bv[h] = (half_t)W[(size_t)kk * Kout + col];   // coalesced across lanes (col)
    }
    acc = __builtin_amdgcn_wmma_f32_16x16x32_f16(false, av, false, bv,
                                                 (short)0, acc, false, false);
  }
  float* cp = C + (size_t)(tileM * 16) * Kout + tileN * 16 + (lane & 15);
  const int mofs = (lane < 16) ? 0 : 8;
#pragma unroll
  for (int r = 0; r < 8; ++r)
    cp[(size_t)(r + mofs) * Kout] = acc[r];
}

// ---------------------------------------------------------------------------
// GCN aggregation: out[dst,f] += h[src,f] * dinv[src]*dinv[dst]  (atomics in L2)
// ---------------------------------------------------------------------------
template <int F>
__global__ void gcn_agg_kernel(const float* __restrict__ h, const int* __restrict__ ei,
                               const float* __restrict__ dinv, float* __restrict__ out) {
  int t = blockIdx.x * blockDim.x + threadIdx.x;
  if (t >= N_EDGES * F) return;
  int e = t / F, f = t - e * F;
  int s = ei[e], d = ei[N_EDGES + e];
  atomicAdd(&out[(size_t)d * F + f], h[(size_t)s * F + f] * dinv[s] * dinv[d]);
}

template <int F>
__global__ void gcn_fin_kernel(const float* __restrict__ agg, const float* __restrict__ hlin,
                               const float* __restrict__ dinv, const float* __restrict__ bias,
                               float* __restrict__ out) {
  int t = blockIdx.x * blockDim.x + threadIdx.x;
  if (t >= N_NODES * F) return;
  int i = t / F, f = t - i * F;
  float di = dinv[i];
  float v = agg[t] + hlin[t] * di * di + bias[f];   // self-loop term fused here
  out[t] = fmaxf(v, 0.0f);
}

// ---------------------------------------------------------------------------
// GAT pieces
// ---------------------------------------------------------------------------
__global__ void gat_scores_kernel(const float* __restrict__ h, const float* __restrict__ asrc,
                                  const float* __restrict__ adst, float* __restrict__ sb,
                                  float* __restrict__ db, int ch) {
  int t = blockIdx.x * blockDim.x + threadIdx.x;
  if (t >= N_NODES * N_HEADS) return;
  int i = t / N_HEADS, hd = t - i * N_HEADS;
  const float* hp = h + (size_t)i * (N_HEADS * ch) + hd * ch;
  const float* as = asrc + hd * ch;
  const float* ad = adst + hd * ch;
  float s = 0.0f, d = 0.0f;
  for (int c = 0; c < ch; ++c) { float x = hp[c]; s += x * as[c]; d += x * ad[c]; }
  sb[t] = s; db[t] = d;
}

__device__ inline void edge_sd(const int* __restrict__ ei, int e, int& s, int& d) {
  if (e < N_EDGES) { s = ei[e]; d = ei[N_EDGES + e]; }
  else             { s = e - N_EDGES; d = s; }            // appended self-loops
}

__device__ inline void atomicMaxF(float* addr, float val) {
  // signed-max / unsigned-min bit trick; works with mixed signs, init = -1e30f
  if (val >= 0.0f) atomicMax((int*)addr, __float_as_int(val));
  else             atomicMin((unsigned int*)addr, (unsigned int)__float_as_int(val));
}

__device__ inline float leaky02(float x) { return x > 0.0f ? x : 0.2f * x; }

__global__ void gat_max_kernel(const int* __restrict__ ei, const float* __restrict__ sb,
                               const float* __restrict__ db, float* __restrict__ mb) {
  int t = blockIdx.x * blockDim.x + threadIdx.x;
  if (t >= (N_EDGES + N_NODES) * N_HEADS) return;
  int e = t / N_HEADS, hd = t - e * N_HEADS;
  int s, d; edge_sd(ei, e, s, d);
  float x = leaky02(sb[s * N_HEADS + hd] + db[d * N_HEADS + hd]);
  atomicMaxF(&mb[d * N_HEADS + hd], x);
}

__global__ void gat_alpha_kernel(const int* __restrict__ ei, const float* __restrict__ sb,
                                 const float* __restrict__ db, const float* __restrict__ mb,
                                 float* __restrict__ zb, float* __restrict__ alpha) {
  int t = blockIdx.x * blockDim.x + threadIdx.x;
  if (t >= (N_EDGES + N_NODES) * N_HEADS) return;
  int e = t / N_HEADS, hd = t - e * N_HEADS;
  int s, d; edge_sd(ei, e, s, d);
  float x = leaky02(sb[s * N_HEADS + hd] + db[d * N_HEADS + hd]);
  float a = __expf(x - mb[d * N_HEADS + hd]);
  alpha[t] = a;
  atomicAdd(&zb[d * N_HEADS + hd], a);
}

template <int F, int CH>
__global__ void gat_agg_kernel(const int* __restrict__ ei, const float* __restrict__ h,
                               const float* __restrict__ alpha, float* __restrict__ out) {
  int t = blockIdx.x * blockDim.x + threadIdx.x;
  if (t >= (N_EDGES + N_NODES) * F) return;
  int e = t / F, f = t - e * F;
  int hd = f / CH;
  int s, d; edge_sd(ei, e, s, d);
  atomicAdd(&out[(size_t)d * F + f], alpha[e * N_HEADS + hd] * h[(size_t)s * F + f]);
}

template <int F, int CH>
__global__ void gat1_fin_kernel(const float* __restrict__ agg, const float* __restrict__ zb,
                                const float* __restrict__ bias, float* __restrict__ out) {
  int t = blockIdx.x * blockDim.x + threadIdx.x;
  if (t >= N_NODES * F) return;
  int i = t / F, f = t - i * F;
  int hd = f / CH;
  float v = agg[t] / (zb[i * N_HEADS + hd] + 1e-16f) + bias[f];
  out[t] = fmaxf(v, 0.0f);   // relu (concat layer)
}

// GAT2 finalize (mean over heads + bias) fused with log_softmax; one wave32 per
// node, lane == class (OUT == 32 == warpSize).
__global__ void gat2_out_kernel(const float* __restrict__ agg, const float* __restrict__ zb,
                                const float* __restrict__ bias, float* __restrict__ out) {
  int lane = threadIdx.x & 31;
  int node = blockIdx.x * (blockDim.x >> 5) + (threadIdx.x >> 5);
  if (node >= N_NODES) return;
  float z0 = zb[node * 2 + 0] + 1e-16f;
  float z1 = zb[node * 2 + 1] + 1e-16f;
  float v = 0.5f * (agg[(size_t)node * 64 + lane] / z0 +
                    agg[(size_t)node * 64 + 32 + lane] / z1) + bias[lane];
  float m = v;
#pragma unroll
  for (int o = 16; o > 0; o >>= 1) m = fmaxf(m, __shfl_xor(m, o));
  float ex = __expf(v - m);
  float sum = ex;
#pragma unroll
  for (int o = 16; o > 0; o >>= 1) sum += __shfl_xor(sum, o);
  out[(size_t)node * 32 + lane] = v - m - logf(sum);
}

// ---------------------------------------------------------------------------
extern "C" void kernel_launch(void* const* d_in, const int* in_sizes, int n_in,
                              void* d_out, int out_size, void* d_ws, size_t ws_size,
                              hipStream_t stream) {
  (void)in_sizes; (void)n_in; (void)out_size; (void)ws_size;
  const float* x   = (const float*)d_in[0];
  const int*   ei  = (const int*)d_in[1];
  const float* W1  = (const float*)d_in[2];
  const float* b1  = (const float*)d_in[3];
  const float* W2  = (const float*)d_in[4];
  const float* b2  = (const float*)d_in[5];
  const float* Wg1 = (const float*)d_in[6];
  const float* as1 = (const float*)d_in[7];
  const float* ad1 = (const float*)d_in[8];
  const float* bg1 = (const float*)d_in[9];
  const float* Wg2 = (const float*)d_in[10];
  const float* as2 = (const float*)d_in[11];
  const float* ad2 = (const float*)d_in[12];
  const float* bg2 = (const float*)d_in[13];
  float* out = (float*)d_out;

  float* ws   = (float*)d_ws;
  float* bufA = ws; ws += (size_t)N_NODES * 128;
  float* bufB = ws; ws += (size_t)N_NODES * 128;
  float* agg  = ws; ws += (size_t)N_NODES * 128;
  float* deg  = ws; ws += N_NODES;
  float* dinv = ws; ws += N_NODES;
  float* sb   = ws; ws += N_NODES * N_HEADS;
  float* db   = ws; ws += N_NODES * N_HEADS;
  float* mb   = ws; ws += N_NODES * N_HEADS;
  float* zb   = ws; ws += N_NODES * N_HEADS;
  float* alpha = ws; ws += (size_t)(N_EDGES + N_NODES) * N_HEADS;

  dim3 blk(256);
  auto g1 = [](long long total) { return dim3((unsigned)((total + 255) / 256)); };
  auto gemm = [&](const float* A, const float* W, float* C, int Kin, int Kout) {
    int waves = (N_NODES / 16) * (Kout / 16);
    wmma_gemm_f16<<<dim3((unsigned)((waves + 3) / 4)), dim3(128), 0, stream>>>(
        A, W, C, N_NODES, Kin, Kout);
  };

  // ---- degrees (self-loop folded in as +1) ----
  hipMemsetAsync(deg, 0, sizeof(float) * N_NODES, stream);
  deg_kernel<<<g1(N_EDGES), blk, 0, stream>>>(ei, deg);
  dinv_kernel<<<g1(N_NODES), blk, 0, stream>>>(deg, dinv);

  // ---- GCN layer 1: x(N,128) @ W1 -> 64, sym-norm aggregate, +b, relu ----
  gemm(x, W1, bufA, DIM_IN, DIM_HID);
  hipMemsetAsync(agg, 0, sizeof(float) * (size_t)N_NODES * DIM_HID, stream);
  gcn_agg_kernel<DIM_HID><<<g1((long long)N_EDGES * DIM_HID), blk, 0, stream>>>(bufA, ei, dinv, agg);
  gcn_fin_kernel<DIM_HID><<<g1((long long)N_NODES * DIM_HID), blk, 0, stream>>>(agg, bufA, dinv, b1, bufB);

  // ---- GCN layer 2: (N,64) @ W2 -> 64 ----
  gemm(bufB, W2, bufA, DIM_HID, DIM_HID);
  hipMemsetAsync(agg, 0, sizeof(float) * (size_t)N_NODES * DIM_HID, stream);
  gcn_agg_kernel<DIM_HID><<<g1((long long)N_EDGES * DIM_HID), blk, 0, stream>>>(bufA, ei, dinv, agg);
  gcn_fin_kernel<DIM_HID><<<g1((long long)N_NODES * DIM_HID), blk, 0, stream>>>(agg, bufA, dinv, b2, bufB);

  // ---- GAT layer 1: (N,64) @ Wg1 -> (N,128) [2 heads x 64], concat, relu ----
  gemm(bufB, Wg1, bufA, DIM_HID, N_HEADS * DIM_HID);
  gat_scores_kernel<<<g1(N_NODES * N_HEADS), blk, 0, stream>>>(bufA, as1, ad1, sb, db, DIM_HID);
  fill_kernel<<<g1(N_NODES * N_HEADS), blk, 0, stream>>>(mb, N_NODES * N_HEADS, -1e30f);
  hipMemsetAsync(zb, 0, sizeof(float) * N_NODES * N_HEADS, stream);
  hipMemsetAsync(agg, 0, sizeof(float) * (size_t)N_NODES * 128, stream);
  gat_max_kernel<<<g1((long long)(N_EDGES + N_NODES) * N_HEADS), blk, 0, stream>>>(ei, sb, db, mb);
  gat_alpha_kernel<<<g1((long long)(N_EDGES + N_NODES) * N_HEADS), blk, 0, stream>>>(ei, sb, db, mb, zb, alpha);
  gat_agg_kernel<128, DIM_HID><<<g1((long long)(N_EDGES + N_NODES) * 128), blk, 0, stream>>>(ei, bufA, alpha, agg);
  gat1_fin_kernel<128, DIM_HID><<<g1((long long)N_NODES * 128), blk, 0, stream>>>(agg, zb, bg1, bufB);

  // ---- GAT layer 2: (N,128) @ Wg2 -> (N,64) [2 heads x 32], mean heads ----
  gemm(bufB, Wg2, bufA, N_HEADS * DIM_HID, N_HEADS * DIM_OUT);
  gat_scores_kernel<<<g1(N_NODES * N_HEADS), blk, 0, stream>>>(bufA, as2, ad2, sb, db, DIM_OUT);
  fill_kernel<<<g1(N_NODES * N_HEADS), blk, 0, stream>>>(mb, N_NODES * N_HEADS, -1e30f);
  hipMemsetAsync(zb, 0, sizeof(float) * N_NODES * N_HEADS, stream);
  hipMemsetAsync(agg, 0, sizeof(float) * (size_t)N_NODES * 64, stream);
  gat_max_kernel<<<g1((long long)(N_EDGES + N_NODES) * N_HEADS), blk, 0, stream>>>(ei, sb, db, mb);
  gat_alpha_kernel<<<g1((long long)(N_EDGES + N_NODES) * N_HEADS), blk, 0, stream>>>(ei, sb, db, mb, zb, alpha);
  gat_agg_kernel<64, DIM_OUT><<<g1((long long)(N_EDGES + N_NODES) * 64), blk, 0, stream>>>(ei, bufA, alpha, agg);

  // ---- finalize + log_softmax (one wave32 per node) ----
  gat2_out_kernel<<<dim3((N_NODES + 7) / 8), dim3(256), 0, stream>>>(agg, zb, bg2, out);
}
